// MGCN_75677323756011
// MI455X (gfx1250) — matrix operand order repaired and checked by reading, not compiled, not run on hardware
//
#include <hip/hip_runtime.h>
#include <hip/hip_bf16.h>

// ---------------------------------------------------------------------------
// CDNA5 WMMA types (wave32): A/B = 16 bf16 per lane, C/D = 8 f32 per lane.
// ---------------------------------------------------------------------------
typedef __attribute__((ext_vector_type(16))) __bf16 v16bf;
typedef __attribute__((ext_vector_type(8)))  float  v8f;

// native f32 -> bf16 (RNE via hardware convert)
static __device__ __forceinline__ __bf16 f32_to_bf16(float f) {
    return static_cast<__bf16>(f);
}
static __device__ __forceinline__ unsigned short f32_to_bf16_raw(float f) {
    return __builtin_bit_cast(unsigned short, static_cast<__bf16>(f));
}

#define BM 128
#define BK 32

// ---------------------------------------------------------------------------
// Fragment index maps (k order as loaded by the kernels below).
// A 16x32:  lanes 0-15 -> M=lane, K=0..15; lanes 16-31 -> K=16..31 (contiguous)
// B 32x16:  lanes 0-15 -> N=lane, K=0..15;         lanes 16-31 -> K=16..31 (order used here, same as A)
// C/D f32:  VGPR v -> M = v + (lane>=16 ? 8 : 0), N = lane&15
// ---------------------------------------------------------------------------
static __device__ __forceinline__ int a_kmap(int j, int hi_half) {
    int kbase = hi_half ? 8 : 0;
    return (j < 8) ? (kbase + j) : (16 + kbase + (j - 8));
}

// ===========================================================================
// Fast-path GEMM for the big layers (no residual fusion, K%32==0):
//   - grid.x = N-tile, grid.y = M-tile: consecutive blocks share the same
//     128-row A slab -> A strip stays L2-resident, ~1x HBM traffic for A
//   - A tile staged with GLOBAL_LOAD_ASYNC_TO_LDS_B128 (ASYNCcnt), f32 in LDS
//   - double-buffered LDS: next tile's async DMA overlaps current WMMAs
//   - RN 16-col subtiles per wave: one A fragment feeds RN WMMAs (A reuse)
// ===========================================================================
template <int RN>
__global__ __launch_bounds__(256)
void gemm_bf16_wmma_async(const float* __restrict__ A, const float* __restrict__ B,
                          float* __restrict__ C, int M, int K, int Ncols)
{
    constexpr int BN = 16 * RN;
    __shared__ float Asf[2][BM * BK];   // 2 x 16 KB, row-major [row][k]
    __shared__ float Bsf[2][BN * BK];   // transposed [n][k]

    const int tid  = threadIdx.x;
    const int wave = tid >> 5;
    const int lane = tid & 31;
    const int tileN = blockIdx.x * BN;
    const int tileM = blockIdx.y * BM;
    const int r16     = lane & 15;
    const int hi_half = lane & 16;

    const unsigned long long Abase = (unsigned long long)A;

    auto stageA = [&](int buf, int k0) {
        unsigned abase = (unsigned)(uintptr_t)(&Asf[buf][0]);
        #pragma unroll
        for (int p = 0; p < 4; ++p) {               // 4096 dwords / 256 thr / 4 dw
            int d   = (p * 256 + tid) * 4;          // dword index in tile
            int row = d >> 5;                        // /BK
            int kk  = d & (BK - 1);
            int gr  = tileM + row; if (gr >= M) gr = M - 1;   // clamp: pad rows
            unsigned voff = ((unsigned)gr * (unsigned)K + (unsigned)(k0 + kk)) * 4u;
            unsigned ldsa = abase + (unsigned)d * 4u;
            asm volatile("global_load_async_to_lds_b128 %0, %1, %2 offset:0"
                         :: "v"(ldsa), "v"(voff), "s"(Abase) : "memory");
        }
    };
    auto stageB = [&](int buf, int k0) {
        constexpr int ITERS = (BN * BK) / 256;      // compile-time trip count
        float tmp[ITERS];
        #pragma unroll
        for (int i = 0; i < ITERS; ++i) {           // batch loads first
            int t  = i * 256 + tid;
            int kk = t / BN;
            int nn = t % BN;
            tmp[i] = B[(long long)(k0 + kk) * Ncols + (tileN + nn)];
        }
        #pragma unroll
        for (int i = 0; i < ITERS; ++i) {           // then stores (transpose)
            int t  = i * 256 + tid;
            int kk = t / BN;
            int nn = t % BN;
            Bsf[buf][nn * BK + kk] = tmp[i];
        }
    };

    v8f acc[RN];
    #pragma unroll
    for (int r = 0; r < RN; ++r) acc[r] = v8f{};

    const int nT = K / BK;
    stageA(0, 0);
    stageB(0, 0);

    for (int it = 0; it < nT; ++it) {
        const int cur = it & 1;
        asm volatile("s_wait_asynccnt 0x0" ::: "memory");  // this wave's DMA done
        __syncthreads();                                   // whole tile visible

        if (it + 1 < nT) {                                 // prefetch next tile
            stageA(cur ^ 1, (it + 1) * BK);
            stageB(cur ^ 1, (it + 1) * BK);
        }

        v16bf a;
        {
            const float* ap = &Asf[cur][(wave * 16 + r16) * BK];
            #pragma unroll
            for (int j = 0; j < 16; ++j)
                a[j] = f32_to_bf16(ap[(hi_half ? 16 : 0) + j]);
        }
        #pragma unroll
        for (int r = 0; r < RN; ++r) {
            v16bf b;
            const float* bp = &Bsf[cur][(r * 16 + r16) * BK + (hi_half ? 16 : 0)];
            #pragma unroll
            for (int j = 0; j < 16; ++j)
                b[j] = f32_to_bf16(bp[j]);
            acc[r] = __builtin_amdgcn_wmma_f32_16x16x32_bf16(
                false, a, false, b, (short)0, acc[r], false, false);
        }
    }

    const int mbase = hi_half ? 8 : 0;
    #pragma unroll
    for (int r = 0; r < RN; ++r) {
        int col = tileN + r * 16 + r16;
        #pragma unroll
        for (int v = 0; v < 8; ++v) {
            int row = tileM + wave * 16 + mbase + v;
            if (row < M) C[(long long)row * Ncols + col] = acc[r][v];
        }
    }
}

// ===========================================================================
// Generic GEMM (small layers): optional A-side fusion (alphaA*A + Aadd),
// arbitrary K / Ncols (zero-padded via branch-free clamp+mask staging).
// ===========================================================================
template <bool HAS_ADD>
__global__ __launch_bounds__(256)
void gemm_bf16_wmma_gen(const float* __restrict__ A, const float* __restrict__ Aadd,
                        float alphaA, const float* __restrict__ B,
                        float* __restrict__ C, int M, int K, int Ncols)
{
    constexpr int BN = 16;
    __shared__ unsigned short As[BM * BK];   // bf16 raw, row-major [row][k]
    __shared__ unsigned short Bs[BN * BK];   // bf16 raw, transposed [n][k]

    const int tid  = threadIdx.x;
    const int wave = tid >> 5;
    const int lane = tid & 31;
    const int tileN = blockIdx.x * BN;
    const int tileM = blockIdx.y * BM;
    const int r16     = lane & 15;
    const int hi_half = lane & 16;

    v8f acc = {};

    for (int k0 = 0; k0 < K; k0 += BK) {
        // branch-free A staging: clamp indices, mask padded elements
        {
            float va[16], vb[16];
            #pragma unroll
            for (int i = 0; i < 16; ++i) {          // 4096 elems / 256 thr
                int t   = i * 256 + tid;
                int row = t >> 5;
                int kk  = t & (BK - 1);
                int gr  = tileM + row; if (gr >= M) gr = M - 1;
                int gk  = k0 + kk;
                float mask = (gk < K) ? 1.f : 0.f;
                int gkc = (gk < K) ? gk : (K - 1);
                long long idx = (long long)gr * K + gkc;
                va[i] = A[idx] * (alphaA * mask);
                if (HAS_ADD) vb[i] = Aadd[idx] * mask;
            }
            #pragma unroll
            for (int i = 0; i < 16; ++i) {
                float v = HAS_ADD ? (va[i] + vb[i]) : va[i];
                As[i * 256 + tid] = f32_to_bf16_raw(v);
            }
        }
        // branch-free B staging (transposed)
        {
            float tb[2];
            #pragma unroll
            for (int i = 0; i < 2; ++i) {           // 512 elems / 256 thr
                int t  = i * 256 + tid;
                int kk = t >> 4;
                int nn = t & 15;
                int gk = k0 + kk, gn = tileN + nn;
                float mask = (gk < K && gn < Ncols) ? 1.f : 0.f;
                int gkc = (gk < K) ? gk : (K - 1);
                int gnc = (gn < Ncols) ? gn : (Ncols - 1);
                tb[i] = B[(long long)gkc * Ncols + gnc] * mask;
            }
            #pragma unroll
            for (int i = 0; i < 2; ++i) {
                int t  = i * 256 + tid;
                int kk = t >> 4;
                int nn = t & 15;
                Bs[nn * BK + kk] = f32_to_bf16_raw(tb[i]);
            }
        }
        __syncthreads();

        v16bf a, b;
        {
            const unsigned short* ap = &As[(wave * 16 + r16) * BK];
            #pragma unroll
            for (int j = 0; j < 16; ++j)
                a[j] = __builtin_bit_cast(__bf16, ap[(hi_half ? 16 : 0) + j]);
        }
        {
            const unsigned short* bp = &Bs[r16 * BK + (hi_half ? 16 : 0)];
            #pragma unroll
            for (int j = 0; j < 16; ++j)
                b[j] = __builtin_bit_cast(__bf16, bp[j]);
        }
        acc = __builtin_amdgcn_wmma_f32_16x16x32_bf16(
            false, a, false, b, (short)0, acc, false, false);
        __syncthreads();
    }

    const int col   = tileN + r16;
    const int mbase = hi_half ? 8 : 0;
    if (col < Ncols) {
        #pragma unroll
        for (int v = 0; v < 8; ++v) {
            int row = tileM + wave * 16 + mbase + v;
            if (row < M) C[(long long)row * Ncols + col] = acc[v];
        }
    }
}

// ---------------------------------------------------------------------------
// SpMM scatter: out[dst, :] += scale * w * H[src, :]   (F % 4 == 0)
// Lanes of a wave cover consecutive f4-chunks of ONE edge -> gathers from one
// source row coalesce, edge-data loads broadcast. H rows are L2-resident
// (<=102 MB << 192 MB L2); f32 atomics resolve in L2.
// ---------------------------------------------------------------------------
__global__ void spmm_scatter(const int* __restrict__ src, const int* __restrict__ dst,
                             const float* __restrict__ ew, const float* __restrict__ H,
                             float* __restrict__ out, long long E, int F, float scale)
{
    const int F4 = F >> 2;
    long long id = (long long)blockIdx.x * blockDim.x + threadIdx.x;
    long long total = E * (long long)F4;
    if (id >= total) return;
    long long e = id / F4;
    int f = (int)(id - e * (long long)F4) << 2;
    int s = src[e], d = dst[e];
    float w = ew[e] * scale;
    const float4 h = *reinterpret_cast<const float4*>(H + (long long)s * F + f);
    float* o = out + (long long)d * F + f;
    unsafeAtomicAdd(o + 0, w * h.x);
    unsafeAtomicAdd(o + 1, w * h.y);
    unsafeAtomicAdd(o + 2, w * h.z);
    unsafeAtomicAdd(o + 3, w * h.w);
}

__global__ void fill_zero(float* __restrict__ p, long long n)
{
    long long i = (long long)blockIdx.x * blockDim.x + threadIdx.x;
    if (i < n) p[i] = 0.f;
}

__global__ void relu_inplace(float* __restrict__ p, long long n)
{
    long long i = (long long)blockIdx.x * blockDim.x + threadIdx.x;
    if (i < n) p[i] = fmaxf(p[i], 0.f);
}

// Fused: h* = relu(h*) in place, c = (h0+h1+h2)/6
__global__ void relu3_combine(float* __restrict__ h0, float* __restrict__ h1,
                              float* __restrict__ h2, float* __restrict__ c,
                              long long n)
{
    long long i = (long long)blockIdx.x * blockDim.x + threadIdx.x;
    if (i >= n) return;
    float a = fmaxf(h0[i], 0.f);
    float b = fmaxf(h1[i], 0.f);
    float d = fmaxf(h2[i], 0.f);
    h0[i] = a; h1[i] = b; h2[i] = d;
    c[i] = (a + b + d) * (1.f / 6.f);
}

// ---------------------------------------------------------------------------
// Host-side schedule
// ---------------------------------------------------------------------------
extern "C" void kernel_launch(void* const* d_in, const int* in_sizes, int n_in,
                              void* d_out, int out_size, void* d_ws, size_t ws_size,
                              hipStream_t stream)
{
    const float* x0   = (const float*)d_in[0];
    const float* x1   = (const float*)d_in[1];
    const float* x2   = (const float*)d_in[2];
    const int*   esrc = (const int*)d_in[3];
    const int*   edst = (const int*)d_in[4];
    const float* ew   = (const float*)d_in[5];
    const float* W00 = (const float*)d_in[6];
    const float* W01 = (const float*)d_in[7];
    const float* W02 = (const float*)d_in[8];
    const float* W03 = (const float*)d_in[9];
    const float* W04 = (const float*)d_in[10];
    const float* W10 = (const float*)d_in[11];
    const float* W11 = (const float*)d_in[12];
    const float* W12 = (const float*)d_in[13];
    const float* W13 = (const float*)d_in[14];
    const float* W20 = (const float*)d_in[15];
    const float* W21 = (const float*)d_in[16];
    const float* W22 = (const float*)d_in[17];
    const float* W23 = (const float*)d_in[18];

    const int       N    = in_sizes[0] / 1024;
    const long long E    = (long long)in_sizes[3];
    const int       Cdim = out_size / N;      // 40
    float* outp = (float*)d_out;

    // workspace layout (f32)
    float* P0 = (float*)d_ws;                 // N x 256  GEMM outputs (reused)
    float* P1 = P0 + (size_t)N * 256;         // N x 256  spmm out branch0
    float* H0 = P1 + (size_t)N * 256;         // N x 32 (reused at 16/8)
    float* H1 = H0 + (size_t)N * 32;
    float* H2 = H1 + (size_t)N * 32;
    float* Cb = H2 + (size_t)N * 32;          // N x 32 (reused at 16/8)

    const unsigned gy = (unsigned)((N + BM - 1) / BM);

    auto gemm = [&](const float* A, const float* Aadd, float alpha,
                    const float* W, float* out, int K, int F) {
        if (!Aadd && (K % BK) == 0 && (F % 16) == 0) {
            if (F % 64 == 0) {
                dim3 g((unsigned)(F / 64), gy);
                gemm_bf16_wmma_async<4><<<g, dim3(256), 0, stream>>>(A, W, out, N, K, F);
            } else if (F % 32 == 0) {
                dim3 g((unsigned)(F / 32), gy);
                gemm_bf16_wmma_async<2><<<g, dim3(256), 0, stream>>>(A, W, out, N, K, F);
            } else {
                dim3 g((unsigned)(F / 16), gy);
                gemm_bf16_wmma_async<1><<<g, dim3(256), 0, stream>>>(A, W, out, N, K, F);
            }
        } else {
            dim3 g((unsigned)((F + 15) / 16), gy);
            if (Aadd)
                gemm_bf16_wmma_gen<true><<<g, dim3(256), 0, stream>>>(A, Aadd, alpha, W, out, N, K, F);
            else
                gemm_bf16_wmma_gen<false><<<g, dim3(256), 0, stream>>>(A, nullptr, alpha, W, out, N, K, F);
        }
    };
    auto zero = [&](float* p, long long n) {
        unsigned b = (unsigned)((n + 255) / 256);
        fill_zero<<<dim3(b), dim3(256), 0, stream>>>(p, n);
    };
    auto spmm = [&](const float* t, float* out, int F, float scale) {
        long long total = E * (long long)(F / 4);
        unsigned b = (unsigned)((total + 255) / 256);
        spmm_scatter<<<dim3(b), dim3(256), 0, stream>>>(esrc, edst, ew, t, out, E, F, scale);
    };
    auto relu = [&](float* p, long long n) {
        unsigned b = (unsigned)((n + 255) / 256);
        relu_inplace<<<dim3(b), dim3(256), 0, stream>>>(p, n);
    };
    auto relu3 = [&](float* h0, float* h1, float* h2, float* c, long long n) {
        unsigned b = (unsigned)((n + 255) / 256);
        relu3_combine<<<dim3(b), dim3(256), 0, stream>>>(h0, h1, h2, c, n);
    };

    // ---------------- level 32 ----------------
    gemm(x0, nullptr, 1.f, W00, P0, 1024, 256);                 // async RN=4
    zero(P1, (long long)N * 256); spmm(P0, P1, 256, 1.f); relu(P1, (long long)N * 256);
    gemm(P1, nullptr, 1.f, W01, P0, 256, 32);                   // async RN=2
    zero(H0, (long long)N * 32);  spmm(P0, H0, 32, 1.f);
    gemm(x1, nullptr, 1.f, W10, P0, 512, 32);                   // async RN=2
    zero(H1, (long long)N * 32);  spmm(P0, H1, 32, 1.f);
    gemm(x2, nullptr, 1.f, W20, P0, 512, 32);                   // async RN=2
    zero(H2, (long long)N * 32);  spmm(P0, H2, 32, 1.f);
    relu3(H0, H1, H2, Cb, (long long)N * 32);

    // ---------------- level 16 ----------------  (A-stage fuses h/2 + c)
    gemm(H0, Cb, 0.5f, W02, P0, 32, 16);
    zero(H0, (long long)N * 16);  spmm(P0, H0, 16, 1.f);
    gemm(H1, Cb, 0.5f, W11, P0, 32, 16);
    zero(H1, (long long)N * 16);  spmm(P0, H1, 16, 1.f);
    gemm(H2, Cb, 0.5f, W21, P0, 32, 16);
    zero(H2, (long long)N * 16);  spmm(P0, H2, 16, 1.f);
    relu3(H0, H1, H2, Cb, (long long)N * 16);

    // ---------------- level 8 ----------------
    gemm(H0, Cb, 0.5f, W03, P0, 16, 8);
    zero(H0, (long long)N * 8);   spmm(P0, H0, 8, 1.f);
    gemm(H1, Cb, 0.5f, W12, P0, 16, 8);
    zero(H1, (long long)N * 8);   spmm(P0, H1, 8, 1.f);
    gemm(H2, Cb, 0.5f, W22, P0, 16, 8);
    zero(H2, (long long)N * 8);   spmm(P0, H2, 8, 1.f);
    relu3(H0, H1, H2, Cb, (long long)N * 8);

    // ---------------- output level (linear act) ----------------
    zero(outp, (long long)N * Cdim);
    gemm(H0, Cb, 0.5f, W04, P0, 8, Cdim); spmm(P0, outp, Cdim, 1.f / 3.f);
    gemm(H1, Cb, 0.5f, W13, P0, 8, Cdim); spmm(P0, outp, Cdim, 1.f / 3.f);
    gemm(H2, Cb, 0.5f, W23, P0, 8, Cdim); spmm(P0, outp, Cdim, 1.f / 3.f);
}
